// TopologyEncoder_50800873177281
// MI455X (gfx1250) — matrix operand order repaired and checked by reading, not compiled, not run on hardware
//
#include <hip/hip_runtime.h>
#include <hip/hip_bf16.h>

#define NN 50000
#define NE 1600000
#define DIM 128
#define NL 3
#define NG 64
#define LN_EPS 1e-5f
#define ND (NN * DIM)

typedef __attribute__((ext_vector_type(16))) _Float16 v16h;
typedef __attribute__((ext_vector_type(8)))  float    v8f;

// --- CDNA5 WMMA fragment index helpers (per CDNA5 ISA layouts) ---
// A-matrix 16x32 f16: lanes 0-15 -> M=0..15 (K base 0), lanes 16-31 -> same M (K base +8)
// VGPR v<4: K = half*8 + 2v + p ; v>=4: K = 16 + half*8 + 2(v-4) + p
__device__ __forceinline__ int xhfrag_off(int node, int d) {
  int rowblock = node >> 4;
  int m        = node & 15;
  int kchunk   = d >> 5;
  int kk       = d & 31;
  int half = (kk >> 3) & 1;
  int v    = ((kk >> 4) & 1) * 4 + ((kk >> 1) & 3);
  int p    = kk & 1;
  int lane = m + (half << 4);
  int j    = (v << 1) | p;
  return ((rowblock * 4 + kchunk) * 32 + lane) * 16 + j;
}

// ------------------------- degree / norm kernels -------------------------
__global__ void k_init_deg(float* degp, float* degn) {
  int i = blockIdx.x * blockDim.x + threadIdx.x;
  if (i < NN) { degp[i] = 1.f; degn[i] = 1.f; }
}

__global__ void k_deg(const int* __restrict__ ei, const float* __restrict__ ew,
                      float* degp, float* degn) {
  int e = blockIdx.x * blockDim.x + threadIdx.x;
  if (e >= NE) return;
  int dst = ei[NE + e];
  float w = ew[e];
  if (w > 0.f)      unsafeAtomicAdd(&degp[dst], w);
  else if (w < 0.f) unsafeAtomicAdd(&degn[dst], -w);
}

__global__ void k_rsqrt(float* dp, float* dn) {
  int i = blockIdx.x * blockDim.x + threadIdx.x;
  if (i < NN) { dp[i] = rsqrtf(dp[i]); dn[i] = rsqrtf(dn[i]); }
}

// -------- pre-swizzle W into B-fragment order (K packed 2/dword, lane=N) --------
__global__ void k_wprep(const float* __restrict__ Wp, const float* __restrict__ Wn,
                        _Float16* __restrict__ wfrag) {
  int tid = blockIdx.x * blockDim.x + threadIdx.x;   // NL*2*8*4*32 = 6144
  if (tid >= NL * 2 * 8 * 4 * 32) return;
  int lane   = tid & 31;
  int frag   = tid >> 5;
  int kchunk = frag & 3;
  int ntile  = (frag >> 2) & 7;
  int branch = (frag >> 5) & 1;
  int l      = frag >> 6;
  const float* W = (branch ? Wn : Wp) + l * DIM * DIM;
  _Float16* out = wfrag + frag * 512 + lane * 16;
  int half = lane >> 4, nl = lane & 15;
  for (int j = 0; j < 16; ++j) {
    int v = j >> 1, p = j & 1;
    int k = kchunk * 32 + ((j >> 3) << 4) + half * 8 + (j & 7);
    int n = ntile * 16 + nl;
    out[j] = (_Float16)W[k * DIM + n];
  }
}

__global__ void k_xinit(const float* __restrict__ x, _Float16* __restrict__ xh) {
  int idx = blockIdx.x * blockDim.x + threadIdx.x;
  if (idx >= ND) return;
  xh[xhfrag_off(idx >> 7, idx & 127)] = (_Float16)x[idx];
}

// ----------------------------- WMMA GEMM -----------------------------
// grid = 3125 rowblocks, 256 threads = 8 waves; wave w does ntile=w for both branches
__global__ void __launch_bounds__(256) k_gemm(const _Float16* __restrict__ xh,
                                              const _Float16* __restrict__ wfrag,
                                              int layer,
                                              float* __restrict__ xwp,
                                              float* __restrict__ xwn) {
  int lane = threadIdx.x & 31;
  int wid  = threadIdx.x >> 5;
  int rb   = blockIdx.x;
  const _Float16* A  = xh + (rb * 4) * 512;
  const _Float16* Bp = wfrag + (((layer * 2 + 0) * 8 + wid) * 4) * 512;
  const _Float16* Bn = wfrag + (((layer * 2 + 1) * 8 + wid) * 4) * 512;
  v8f c0 = {}; v8f c1 = {};
  for (int kc = 0; kc < 4; ++kc) {
    v16h a  = *(const v16h*)(A  + kc * 512 + lane * 16);
    v16h b0 = *(const v16h*)(Bp + kc * 512 + lane * 16);
    v16h b1 = *(const v16h*)(Bn + kc * 512 + lane * 16);
    c0 = __builtin_amdgcn_wmma_f32_16x16x32_f16(false, a, false, b0, (short)0, c0, false, false);
    c1 = __builtin_amdgcn_wmma_f32_16x16x32_f16(false, a, false, b1, (short)0, c1, false, false);
  }
  int col  = wid * 16 + (lane & 15);
  int row0 = rb * 16 + ((lane >> 4) << 3);
  for (int r = 0; r < 8; ++r) {
    xwp[(row0 + r) * DIM + col] = c0[r];
    xwn[(row0 + r) * DIM + col] = c1[r];
  }
}

// --------------------- edge scatter: 1 wave per edge ---------------------
__global__ void __launch_bounds__(256) k_scatter(const int* __restrict__ ei,
                                                 const float* __restrict__ ew,
                                                 const float* __restrict__ dinvp,
                                                 const float* __restrict__ dinvn,
                                                 const float* __restrict__ xwp,
                                                 const float* __restrict__ xwn,
                                                 float* __restrict__ aggp,
                                                 float* __restrict__ aggn) {
  int e = blockIdx.x * 8 + (threadIdx.x >> 5);
  if (e >= NE) return;
  int lane = threadIdx.x & 31;
  float w = ew[e];
  if (w == 0.f) return;
  int src = ei[e];
  int dst = ei[NE + e];
  if (w > 0.f) {
    float norm = dinvp[src] * w * dinvp[dst];
    const float4 xv = *(const float4*)(xwp + src * DIM + lane * 4);
    float* b = aggp + dst * DIM + lane * 4;
    unsafeAtomicAdd(b + 0, norm * xv.x);
    unsafeAtomicAdd(b + 1, norm * xv.y);
    unsafeAtomicAdd(b + 2, norm * xv.z);
    unsafeAtomicAdd(b + 3, norm * xv.w);
  } else {
    float norm = dinvn[src] * (-w) * dinvn[dst];
    const float4 xv = *(const float4*)(xwn + src * DIM + lane * 4);
    float* b = aggn + dst * DIM + lane * 4;
    unsafeAtomicAdd(b + 0, norm * xv.x);
    unsafeAtomicAdd(b + 1, norm * xv.y);
    unsafeAtomicAdd(b + 2, norm * xv.z);
    unsafeAtomicAdd(b + 3, norm * xv.w);
  }
}

// ------------- self-loop + bias + relu-diff; feed next layer or pool -------------
__global__ void k_update(const float* __restrict__ aggp, const float* __restrict__ aggn,
                         const float* __restrict__ xwp,  const float* __restrict__ xwn,
                         const float* __restrict__ dinvp, const float* __restrict__ dinvn,
                         const float* __restrict__ bp, const float* __restrict__ bn,
                         int layer, int last,
                         _Float16* __restrict__ xh_next,
                         const int* __restrict__ batch, float* __restrict__ sums) {
  int idx = blockIdx.x * blockDim.x + threadIdx.x;
  if (idx >= ND) return;
  int node = idx >> 7, d = idx & 127;
  float dp = dinvp[node], dn = dinvn[node];
  float px = aggp[idx] + dp * dp * xwp[idx] + bp[layer * DIM + d];
  float nx = aggn[idx] + dn * dn * xwn[idx] + bn[layer * DIM + d];
  float val = fmaxf(px, 0.f) - fmaxf(nx, 0.f);
  if (last) {
    unsafeAtomicAdd(&sums[batch[node] * DIM + d], val);
  } else {
    xh_next[xhfrag_off(node, d)] = (_Float16)val;
  }
}

__global__ void k_counts(const int* __restrict__ batch, float* __restrict__ counts) {
  int i = blockIdx.x * blockDim.x + threadIdx.x;
  if (i < NN) unsafeAtomicAdd(&counts[batch[i]], 1.f);
}

// ----------------------------- mean-pool + LayerNorm -----------------------------
__global__ void __launch_bounds__(128) k_ln(const float* __restrict__ sums,
                                            const float* __restrict__ counts,
                                            const float* __restrict__ gamma,
                                            const float* __restrict__ beta,
                                            float* __restrict__ out) {
  __shared__ float red[128];
  int g = blockIdx.x, d = threadIdx.x;
  float cnt = fmaxf(counts[g], 1.f);
  float p = sums[g * DIM + d] / cnt;
  red[d] = p; __syncthreads();
  for (int s = 64; s > 0; s >>= 1) { if (d < s) red[d] += red[d + s]; __syncthreads(); }
  float mu = red[0] / (float)DIM;
  __syncthreads();
  float diff = p - mu;
  red[d] = diff * diff; __syncthreads();
  for (int s = 64; s > 0; s >>= 1) { if (d < s) red[d] += red[d + s]; __syncthreads(); }
  float var = red[0] / (float)DIM;
  out[g * DIM + d] = diff * rsqrtf(var + LN_EPS) * gamma[d] + beta[d];
}

extern "C" void kernel_launch(void* const* d_in, const int* in_sizes, int n_in,
                              void* d_out, int out_size, void* d_ws, size_t ws_size,
                              hipStream_t stream) {
  const float* x     = (const float*)d_in[0];
  const int*   ei    = (const int*)d_in[1];
  const float* ew    = (const float*)d_in[2];
  const int*   batch = (const int*)d_in[3];
  const float* Wp    = (const float*)d_in[4];
  const float* bp    = (const float*)d_in[5];
  const float* Wn    = (const float*)d_in[6];
  const float* bn    = (const float*)d_in[7];
  const float* gamma = (const float*)d_in[8];
  const float* beta  = (const float*)d_in[9];
  float* out = (float*)d_out;

  size_t off = 0;
  char* base = (char*)d_ws;
  auto alloc = [&](size_t bytes) -> void* {
    void* p = base + off;
    off = (off + bytes + 255) & ~(size_t)255;
    return p;
  };
  float*    xwp    = (float*)alloc((size_t)ND * 4);
  float*    xwn    = (float*)alloc((size_t)ND * 4);
  float*    aggp   = (float*)alloc((size_t)ND * 4);
  float*    aggn   = (float*)alloc((size_t)ND * 4);
  _Float16* xh     = (_Float16*)alloc((size_t)ND * 2);
  _Float16* wfrag  = (_Float16*)alloc((size_t)NL * 2 * DIM * DIM * 2);
  float*    dinvp  = (float*)alloc((size_t)NN * 4);
  float*    dinvn  = (float*)alloc((size_t)NN * 4);
  float*    sums   = (float*)alloc((size_t)NG * DIM * 4);
  float*    counts = (float*)alloc((size_t)NG * 4);

  hipMemsetAsync(sums, 0, (size_t)NG * DIM * 4, stream);
  hipMemsetAsync(counts, 0, (size_t)NG * 4, stream);

  k_init_deg<<<(NN + 255) / 256, 256, 0, stream>>>(dinvp, dinvn);
  k_deg<<<(NE + 255) / 256, 256, 0, stream>>>(ei, ew, dinvp, dinvn);
  k_rsqrt<<<(NN + 255) / 256, 256, 0, stream>>>(dinvp, dinvn);
  k_wprep<<<(NL * 2 * 8 * 4 * 32 + 255) / 256, 256, 0, stream>>>(Wp, Wn, wfrag);
  k_xinit<<<(ND + 255) / 256, 256, 0, stream>>>(x, xh);
  k_counts<<<(NN + 255) / 256, 256, 0, stream>>>(batch, counts);

  for (int l = 0; l < NL; ++l) {
    k_gemm<<<NN / 16, 256, 0, stream>>>(xh, wfrag, l, xwp, xwn);
    hipMemsetAsync(aggp, 0, (size_t)ND * 4, stream);
    hipMemsetAsync(aggn, 0, (size_t)ND * 4, stream);
    k_scatter<<<(NE + 7) / 8, 256, 0, stream>>>(ei, ew, dinvp, dinvn, xwp, xwn, aggp, aggn);
    k_update<<<(ND + 255) / 256, 256, 0, stream>>>(aggp, aggn, xwp, xwn, dinvp, dinvn,
                                                   bp, bn, l, (l == NL - 1) ? 1 : 0,
                                                   xh, batch, sums);
  }
  k_ln<<<NG, 128, 0, stream>>>(sums, counts, gamma, beta, out);
}
